// CSPNet_266287972901
// MI455X (gfx1250) — compile-verified, matrix operand order checked by reading, not run on hardware
//
#include <hip/hip_runtime.h>
#include <hip/hip_bf16.h>
#include <stdint.h>

// ---------------------------------------------------------------------------
// Types for CDNA5 WMMA (wave32): v16bf A/B fragments, v8f C/D accumulator.
// ---------------------------------------------------------------------------
typedef __bf16 bf16;
typedef bf16  v16bf __attribute__((ext_vector_type(16)));
typedef bf16  v8bf  __attribute__((ext_vector_type(8)));
typedef bf16  v4bf  __attribute__((ext_vector_type(4)));
typedef float v8f   __attribute__((ext_vector_type(8)));

__device__ __forceinline__ float siluf(float x) {
    return x / (1.0f + __expf(-x));
}

// Load a 16x32 bf16 fragment from a row-major matrix with row stride `stride`
// (elements). Matches CDNA5 16-bit A-matrix layout:
//   lane<16 : row=lane,    K = {0..7} (elems 0-7), {16..23} (elems 8-15)
//   lane>=16: row=lane-16, K = {8..15},            {24..31}
// B fragments use the same pattern on an N x K (row-major) weight matrix.
__device__ __forceinline__ v16bf load_frag_rm(const bf16* base, int stride) {
    const int lane = threadIdx.x & 31;
    const int r    = lane & 15;
    const int kb   = (lane >> 4) << 3;      // 0 or 8
    const bf16* q  = base + (size_t)r * stride + kb;
    v8bf lo = *(const v8bf*)(q);
    v8bf hi = *(const v8bf*)(q + 16);
    v16bf f;
#pragma unroll
    for (int i = 0; i < 8; ++i) { f[i] = lo[i]; f[i + 8] = hi[i]; }
    return f;
}

__device__ __forceinline__ v8f wmma_bf16(v16bf a, v16bf b, v8f c) {
    return __builtin_amdgcn_wmma_f32_16x16x32_bf16(
        false, a, false, b, (short)0, c, false, false);
}

__device__ __forceinline__ v8f zero8() {
    v8f z;
#pragma unroll
    for (int i = 0; i < 8; ++i) z[i] = 0.0f;
    return z;
}

// ---------------------------------------------------------------------------
// Generic WMMA GEMM: out = act(A @ Bw^T + bias)
//   A  : M x K bf16, row stride lda (K padded to 32)
//   Bw : N x K bf16, row stride lda (weights, pre-packed)
//   outF (fp32) and/or outB (bf16), both M x N row-major.
// Block = 256 threads = 8 waves = 32x64 output tile. Grid = (M/32, N/64).
// ---------------------------------------------------------------------------
__global__ __launch_bounds__(256) void gemm_wmma_kernel(
    const bf16* __restrict__ A, int lda, const bf16* __restrict__ Bw,
    const float* __restrict__ bias, float* __restrict__ outF,
    bf16* __restrict__ outB, int N, int K, int act)
{
    const int wv   = threadIdx.x >> 5;
    const int mi   = wv >> 2, ni = wv & 3;
    const int row0 = blockIdx.x * 32 + mi * 16;
    const int col0 = blockIdx.y * 64 + ni * 16;

    v8f acc = zero8();
    for (int k = 0; k < K; k += 32) {
        v16bf a = load_frag_rm(A  + (size_t)row0 * lda + k, lda);
        v16bf b = load_frag_rm(Bw + (size_t)col0 * lda + k, lda);
        acc = wmma_bf16(a, b, acc);
    }

    const int lane = threadIdx.x & 31;
    const int col  = col0 + (lane & 15);
    const int rb   = (lane >> 4) * 8;
    const float bv = bias ? bias[col] : 0.0f;
#pragma unroll
    for (int r = 0; r < 8; ++r) {
        float x = acc[r] + bv;
        if (act) x = siluf(x);
        size_t o = (size_t)(row0 + rb + r) * N + col;
        if (outF) outF[o] = x;
        if (outB) outB[o] = (bf16)x;
    }
}

// ---------------------------------------------------------------------------
// Fused edge MLP + segment-sum. One block per (graph b, source atoms i0,i0+1):
// 48 edges = exactly 3 M-tiles (no padding).
// ein = [nf[e0](256) | nf[e1](256) | lat(9)+demb(60) pad to 96], K1=608, K2=256.
// agg[node] = sum_j silu(silu(ein@W1^T+b1)@W2^T+b2); rows 0-23 -> i0,
// rows 24-47 -> i0+1.
// ---------------------------------------------------------------------------
__global__ __launch_bounds__(256) void edge_mlp_agg_kernel(
    const bf16* __restrict__ nf_bf, const bf16* __restrict__ tail_bf,
    const bf16* __restrict__ W1, const float* __restrict__ b1,
    const bf16* __restrict__ W2, const float* __restrict__ b2,
    float* __restrict__ agg)
{
    __shared__ bf16  ein[48 * 608];   // 58368 B
    __shared__ bf16  ef1[48 * 256];   // 24576 B
    __shared__ float lagg[2][256];    //  2048 B

    const int tid  = threadIdx.x;
    const int pair = blockIdx.x;          // 0 .. 128*12-1
    const int b    = pair / 12;
    const int i0   = (pair % 12) * 2;

    lagg[0][tid] = 0.0f;
    lagg[1][tid] = 0.0f;

    // Build ein (48 rows x 608 cols), 4 bf16 at a time. All rows are real.
    for (int idx = tid; idx < 48 * 152; idx += 256) {
        const int row = idx / 152;
        const int c   = (idx % 152) * 4;
        const int hi  = (row >= 24) ? 1 : 0;
        const int i   = i0 + hi;
        const int j   = row - hi * 24;
        v4bf v;
        if (c < 256) {
            v = *(const v4bf*)(nf_bf + (size_t)(b * 24 + i) * 256 + c);
        } else if (c < 512) {
            v = *(const v4bf*)(nf_bf + (size_t)(b * 24 + j) * 256 + (c - 256));
        } else {
            v = *(const v4bf*)(tail_bf + (size_t)(b * 576 + i * 24 + j) * 96 + (c - 512));
        }
        *(v4bf*)(ein + row * 608 + c) = v;
    }
    __syncthreads();

    const int wv   = tid >> 5;            // wave -> cols [wv*32, wv*32+32)
    const int lane = tid & 31;
    const int col  = lane & 15;
    const int rb   = (lane >> 4) * 8;

    // ---- stage 1: ef1 = silu(ein @ W1^T + b1), K = 608 ----
    {
        v8f acc[3][2];
#pragma unroll
        for (int mt = 0; mt < 3; ++mt)
#pragma unroll
            for (int nt = 0; nt < 2; ++nt) acc[mt][nt] = zero8();

        for (int k = 0; k < 608; k += 32) {
            v16bf a0 = load_frag_rm(ein + k, 608);
            v16bf a1 = load_frag_rm(ein + 16 * 608 + k, 608);
            v16bf a2 = load_frag_rm(ein + 32 * 608 + k, 608);
            v16bf w0 = load_frag_rm(W1 + (size_t)(wv * 32)      * 608 + k, 608);
            v16bf w1 = load_frag_rm(W1 + (size_t)(wv * 32 + 16) * 608 + k, 608);
            acc[0][0] = wmma_bf16(a0, w0, acc[0][0]);
            acc[0][1] = wmma_bf16(a0, w1, acc[0][1]);
            acc[1][0] = wmma_bf16(a1, w0, acc[1][0]);
            acc[1][1] = wmma_bf16(a1, w1, acc[1][1]);
            acc[2][0] = wmma_bf16(a2, w0, acc[2][0]);
            acc[2][1] = wmma_bf16(a2, w1, acc[2][1]);
        }
#pragma unroll
        for (int mt = 0; mt < 3; ++mt)
#pragma unroll
            for (int nt = 0; nt < 2; ++nt) {
                const int cc = wv * 32 + nt * 16 + col;
                const float bv = b1[cc];
#pragma unroll
                for (int r = 0; r < 8; ++r) {
                    float x = siluf(acc[mt][nt][r] + bv);
                    ef1[(mt * 16 + rb + r) * 256 + cc] = (bf16)x;
                }
            }
    }
    __syncthreads();

    // ---- stage 2: ef2 = silu(ef1 @ W2^T + b2), K = 256; per-node row sum ----
    {
        v8f acc[3][2];
#pragma unroll
        for (int mt = 0; mt < 3; ++mt)
#pragma unroll
            for (int nt = 0; nt < 2; ++nt) acc[mt][nt] = zero8();

        for (int k = 0; k < 256; k += 32) {
            v16bf a0 = load_frag_rm(ef1 + k, 256);
            v16bf a1 = load_frag_rm(ef1 + 16 * 256 + k, 256);
            v16bf a2 = load_frag_rm(ef1 + 32 * 256 + k, 256);
            v16bf w0 = load_frag_rm(W2 + (size_t)(wv * 32)      * 256 + k, 256);
            v16bf w1 = load_frag_rm(W2 + (size_t)(wv * 32 + 16) * 256 + k, 256);
            acc[0][0] = wmma_bf16(a0, w0, acc[0][0]);
            acc[0][1] = wmma_bf16(a0, w1, acc[0][1]);
            acc[1][0] = wmma_bf16(a1, w0, acc[1][0]);
            acc[1][1] = wmma_bf16(a1, w1, acc[1][1]);
            acc[2][0] = wmma_bf16(a2, w0, acc[2][0]);
            acc[2][1] = wmma_bf16(a2, w1, acc[2][1]);
        }
#pragma unroll
        for (int mt = 0; mt < 3; ++mt)
#pragma unroll
            for (int nt = 0; nt < 2; ++nt) {
                const int cc = wv * 32 + nt * 16 + col;
                const float bv = b2[cc];
                float s0 = 0.0f, s1 = 0.0f;
#pragma unroll
                for (int r = 0; r < 8; ++r) {
                    const int row = mt * 16 + rb + r;
                    const float x = siluf(acc[mt][nt][r] + bv);
                    if (row < 24) s0 += x; else s1 += x;
                }
                if (mt == 0) {
                    atomicAdd(&lagg[0][cc], s0);          // ds_add_f32
                } else if (mt == 2) {
                    atomicAdd(&lagg[1][cc], s1);
                } else {
                    atomicAdd(&lagg[0][cc], s0);
                    atomicAdd(&lagg[1][cc], s1);
                }
            }
    }
    __syncthreads();
    agg[(size_t)(b * 24 + i0)     * 256 + tid] = lagg[0][tid];
    agg[(size_t)(b * 24 + i0 + 1) * 256 + tid] = lagg[1][tid];
}

// ---------------------------------------------------------------------------
// LayerNorm + FiLM + residual: nf += silu(LN(y)*scale + shift); also bf16 copy
// ---------------------------------------------------------------------------
__global__ __launch_bounds__(256) void film_kernel(
    const float* __restrict__ y, float* __restrict__ nf, bf16* __restrict__ nf_bf,
    const float* __restrict__ cond, const float* __restrict__ film_g,
    const float* __restrict__ film_b)
{
    __shared__ float red[256];
    const int n = blockIdx.x, tid = threadIdx.x;
    const int g = n / 24;
    const float v = y[(size_t)n * 256 + tid];

    red[tid] = v;
    __syncthreads();
    for (int s = 128; s > 0; s >>= 1) {
        if (tid < s) red[tid] += red[tid + s];
        __syncthreads();
    }
    const float mu = red[0] * (1.0f / 256.0f);
    __syncthreads();
    const float d = v - mu;
    red[tid] = d * d;
    __syncthreads();
    for (int s = 128; s > 0; s >>= 1) {
        if (tid < s) red[tid] += red[tid + s];
        __syncthreads();
    }
    const float var = red[0] * (1.0f / 256.0f);

    const float t  = d * rsqrtf(var + 1e-5f) * film_g[tid] + film_b[tid];
    const float z  = t * cond[(size_t)g * 512 + tid] + cond[(size_t)g * 512 + 256 + tid];
    const float nv = nf[(size_t)n * 256 + tid] + siluf(z);
    nf[(size_t)n * 256 + tid]    = nv;
    nf_bf[(size_t)n * 256 + tid] = (bf16)nv;
}

// cond = silu([t, text] @ w_cond^T + b_cond), one block per graph
__global__ __launch_bounds__(256) void cond_kernel(
    const float* __restrict__ t, const float* __restrict__ te,
    const float* __restrict__ w, const float* __restrict__ bias,
    float* __restrict__ cond)
{
    __shared__ float xin[384];
    const int b = blockIdx.x, tid = threadIdx.x;
    if (tid < 256) xin[tid] = t[b * 256 + tid];
    if (tid < 128) xin[256 + tid] = te[b * 128 + tid];
    __syncthreads();
    for (int o = tid; o < 512; o += 256) {
        float s = bias[o];
        for (int k = 0; k < 384; ++k) s += xin[k] * w[o * 384 + k];
        cond[b * 512 + o] = siluf(s);
    }
}

// lat_ips[b] = L L^T (flattened 9)
__global__ void latips_kernel(const float* __restrict__ L, float* __restrict__ lat_ips)
{
    const int b = blockIdx.x, tid = threadIdx.x;
    if (tid < 9) {
        const int i = tid / 3, k = tid % 3;
        float s = 0.0f;
        for (int j = 0; j < 3; ++j)
            s += L[b * 9 + i * 3 + j] * L[b * 9 + k * 3 + j];
        lat_ips[b * 9 + tid] = s;
    }
}

// Per-edge tail: [lat_ips(9) | sin(30) | cos(30) | zero-pad(27)] in bf16
__global__ void tail_kernel(const float* __restrict__ frac,
                            const float* __restrict__ lat_ips,
                            bf16* __restrict__ tail)
{
    const int idx = blockIdx.x * 256 + threadIdx.x;
    if (idx >= 73728 * 96) return;
    const int e = idx / 96, c = idx % 96;
    const int b = e / 576, rem = e % 576, i = rem / 24, j = rem % 24;
    float v = 0.0f;
    if (c < 9) {
        v = lat_ips[b * 9 + c];
    } else if (c < 69) {
        const int f  = c - 9;
        const int k  = (f < 30) ? f : f - 30;
        const int d  = k / 10, fr = k % 10;
        const int e0 = b * 24 + i, e1 = b * 24 + j;
        float diff = frac[e1 * 3 + d] - frac[e0 * 3 + d];
        diff -= floorf(diff);
        const float ang = diff * (6.283185307179586f * (float)fr);
        v = (f < 30) ? __sinf(ang) : __cosf(ang);
    }
    tail[idx] = (bf16)v;
}

// fp32 -> bf16 with K padding
__global__ void convert_pad_kernel(const float* __restrict__ src, bf16* __restrict__ dst,
                                   int rows, int srcC, int dstC)
{
    const int idx = blockIdx.x * 256 + threadIdx.x;
    if (idx >= rows * dstC) return;
    const int r = idx / dstC, c = idx % dstC;
    dst[idx] = (bf16)((c < srcC) ? src[(size_t)r * srcC + c] : 0.0f);
}

// nfagg = [bf16(nf) | bf16(agg/24)], per node
__global__ __launch_bounds__(256) void pack_nfagg_kernel(
    const bf16* __restrict__ nf_bf, const float* __restrict__ agg,
    bf16* __restrict__ nfagg)
{
    const int idx = blockIdx.x * 256 + threadIdx.x;   // 3072*256
    const int n = idx / 256, c = idx % 256;
    nfagg[(size_t)n * 512 + c]       = nf_bf[idx];
    nfagg[(size_t)n * 512 + 256 + c] = (bf16)(agg[idx] * (1.0f / 24.0f));
}

__global__ __launch_bounds__(256) void node_update_kernel(
    float* __restrict__ nf, const float* __restrict__ no, bf16* __restrict__ nf_bf)
{
    const int idx = blockIdx.x * 256 + threadIdx.x;
    const float v = nf[idx] + no[idx];
    nf[idx]    = v;
    nf_bf[idx] = (bf16)v;
}

// Small output heads: out[n] = nf[n] @ W^T (+ bias), Nout in {3, 103}
__global__ __launch_bounds__(128) void head_kernel(
    const float* __restrict__ nf, const float* __restrict__ W,
    const float* __restrict__ bias, float* __restrict__ out, int Nout)
{
    __shared__ float row[256];
    const int n = blockIdx.x, tid = threadIdx.x;
    row[tid]       = nf[(size_t)n * 256 + tid];
    row[tid + 128] = nf[(size_t)n * 256 + 128 + tid];
    __syncthreads();
    for (int o = tid; o < Nout; o += 128) {
        float s = bias ? bias[o] : 0.0f;
        for (int k = 0; k < 256; ++k) s += row[k] * W[o * 256 + k];
        out[(size_t)n * Nout + o] = s;
    }
}

__global__ __launch_bounds__(256) void gfeat_kernel(const float* __restrict__ nf,
                                                    float* __restrict__ gfeat)
{
    const int b = blockIdx.x, tid = threadIdx.x;
    float s = 0.0f;
    for (int i = 0; i < 24; ++i) s += nf[((size_t)b * 24 + i) * 256 + tid];
    gfeat[b * 256 + tid] = s * (1.0f / 24.0f);
}

__global__ void latt_kernel(const float* __restrict__ gfeat,
                            const float* __restrict__ w_latt,
                            const float* __restrict__ lattices,
                            float* __restrict__ out)
{
    __shared__ float lo[9];
    const int b = blockIdx.x, tid = threadIdx.x;
    if (tid < 9) {
        float s = 0.0f;
        for (int k = 0; k < 256; ++k) s += gfeat[b * 256 + k] * w_latt[tid * 256 + k];
        lo[tid] = s;
    }
    __syncthreads();
    if (tid < 9) {
        const int i = tid / 3, kk = tid % 3;
        float s = 0.0f;
        for (int j = 0; j < 3; ++j) s += lo[i * 3 + j] * lattices[b * 9 + j * 3 + kk];
        out[b * 9 + tid] = s;
    }
}

// ---------------------------------------------------------------------------
extern "C" void kernel_launch(void* const* d_in, const int* in_sizes, int n_in,
                              void* d_out, int out_size, void* d_ws, size_t ws_size,
                              hipStream_t stream)
{
    (void)in_sizes; (void)n_in; (void)out_size; (void)ws_size;
    const float* atom_types = (const float*)d_in[0];
    const float* frac       = (const float*)d_in[1];
    const float* lattices   = (const float*)d_in[2];
    const float* t_in       = (const float*)d_in[3];
    const float* text       = (const float*)d_in[4];
    const float* w_emb      = (const float*)d_in[5];
    const float* b_emb      = (const float*)d_in[6];
    const float* w_cond     = (const float*)d_in[7];
    const float* b_cond     = (const float*)d_in[8];
    const float* w_proj     = (const float*)d_in[9];
    const float* b_proj     = (const float*)d_in[10];
    const float* film_g     = (const float*)d_in[11];
    const float* film_b     = (const float*)d_in[12];
    const float* edge_w1    = (const float*)d_in[13];
    const float* edge_b1    = (const float*)d_in[14];
    const float* edge_w2    = (const float*)d_in[15];
    const float* edge_b2    = (const float*)d_in[16];
    const float* node_w1    = (const float*)d_in[17];
    const float* node_b1    = (const float*)d_in[18];
    const float* node_w2    = (const float*)d_in[19];
    const float* node_b2    = (const float*)d_in[20];
    const float* w_coord    = (const float*)d_in[21];
    const float* w_latt     = (const float*)d_in[22];
    const float* w_type     = (const float*)d_in[23];
    const float* b_type     = (const float*)d_in[24];

    uint8_t* base = (uint8_t*)d_ws;
    size_t off = 0;
    auto alloc = [&](size_t bytes) -> void* {
        void* p = base + off;
        off = (off + bytes + 255) & ~(size_t)255;
        return p;
    };
    float* nf       = (float*)alloc(3072 * 256 * 4);
    float* T0       = (float*)alloc(3072 * 256 * 4);
    float* T1       = (float*)alloc(3072 * 256 * 4);
    float* agg      = (float*)alloc(3072 * 256 * 4);
    float* cond     = (float*)alloc(128 * 512 * 4);
    float* lat_ips  = (float*)alloc(128 * 9 * 4);
    float* gfeat    = (float*)alloc(128 * 256 * 4);
    bf16*  nf_bf    = (bf16*)alloc(3072 * 256 * 2);
    bf16*  tmp_bf   = (bf16*)alloc(3072 * 256 * 2);
    bf16*  nfagg_bf = (bf16*)alloc(3072 * 512 * 2);
    bf16*  atom_bf  = (bf16*)alloc(3072 * 128 * 2);
    bf16*  tail_bf  = (bf16*)alloc((size_t)73728 * 96 * 2);
    bf16*  wemb_bf  = (bf16*)alloc(256 * 128 * 2);
    bf16*  wproj_bf = (bf16*)alloc(256 * 256 * 2);
    bf16*  w1_bf    = (bf16*)alloc((size_t)4 * 256 * 608 * 2);
    bf16*  w2_bf    = (bf16*)alloc((size_t)4 * 256 * 256 * 2);
    bf16*  nw1_bf   = (bf16*)alloc((size_t)4 * 256 * 512 * 2);
    bf16*  nw2_bf   = (bf16*)alloc((size_t)4 * 256 * 256 * 2);

    auto cvt = [&](const float* src, bf16* dst, int rows, int sc, int dc) {
        const int n = rows * dc;
        convert_pad_kernel<<<(n + 255) / 256, 256, 0, stream>>>(src, dst, rows, sc, dc);
    };
    cvt(atom_types, atom_bf,  3072, 103, 128);
    cvt(w_emb,      wemb_bf,   256, 103, 128);
    cvt(w_proj,     wproj_bf,  256, 256, 256);
    cvt(edge_w1,    w1_bf,  4 * 256, 581, 608);
    cvt(edge_w2,    w2_bf,  4 * 256, 256, 256);
    cvt(node_w1,    nw1_bf, 4 * 256, 512, 512);
    cvt(node_w2,    nw2_bf, 4 * 256, 256, 256);

    cond_kernel<<<128, 256, 0, stream>>>(t_in, text, w_cond, b_cond, cond);
    latips_kernel<<<128, 32, 0, stream>>>(lattices, lat_ips);
    tail_kernel<<<(73728 * 96 + 255) / 256, 256, 0, stream>>>(frac, lat_ips, tail_bf);

    // nf = atom_types @ w_emb^T + b_emb   (M=3072, N=256, K=128)
    gemm_wmma_kernel<<<dim3(96, 4), 256, 0, stream>>>(
        atom_bf, 128, wemb_bf, b_emb, nf, nf_bf, 256, 128, 0);

    for (int L = 0; L < 4; ++L) {
        // y = nf @ w_proj^T + b_proj
        gemm_wmma_kernel<<<dim3(96, 4), 256, 0, stream>>>(
            nf_bf, 256, wproj_bf, b_proj, T0, nullptr, 256, 256, 0);
        film_kernel<<<3072, 256, 0, stream>>>(T0, nf, nf_bf, cond, film_g, film_b);

        edge_mlp_agg_kernel<<<1536, 256, 0, stream>>>(
            nf_bf, tail_bf,
            w1_bf + (size_t)L * 256 * 608, edge_b1 + L * 256,
            w2_bf + (size_t)L * 256 * 256, edge_b2 + L * 256, agg);

        pack_nfagg_kernel<<<3072, 256, 0, stream>>>(nf_bf, agg, nfagg_bf);

        gemm_wmma_kernel<<<dim3(96, 4), 256, 0, stream>>>(
            nfagg_bf, 512, nw1_bf + (size_t)L * 256 * 512,
            node_b1 + L * 256, nullptr, tmp_bf, 256, 512, 1);
        gemm_wmma_kernel<<<dim3(96, 4), 256, 0, stream>>>(
            tmp_bf, 256, nw2_bf + (size_t)L * 256 * 256,
            node_b2 + L * 256, T1, nullptr, 256, 256, 1);
        node_update_kernel<<<3072, 256, 0, stream>>>(nf, T1, nf_bf);
    }

    // Outputs: [atom(3072x103) | lattice(128x9) | coords(3072x3) | nf(3072x256)]
    float* out = (float*)d_out;
    head_kernel<<<3072, 128, 0, stream>>>(nf, w_type, b_type, out, 103);
    head_kernel<<<3072, 128, 0, stream>>>(nf, w_coord, nullptr, out + 317568, 3);
    gfeat_kernel<<<128, 256, 0, stream>>>(nf, gfeat);
    latt_kernel<<<128, 32, 0, stream>>>(gfeat, w_latt, lattices, out + 316416);
    hipMemcpyAsync(out + 326784, nf, (size_t)3072 * 256 * sizeof(float),
                   hipMemcpyDeviceToDevice, stream);
}
